// Encoder_6914897346576
// MI455X (gfx1250) — matrix operand order repaired and checked by reading, not compiled, not run on hardware
//
#include <hip/hip_runtime.h>
#include <stdint.h>

// ---------------------------------------------------------------------------
// Persistent bf16-WMMA LSTM encoder for MI455X (gfx1250, wave32).
//   gates[m][n] = ox[m]*Weff0[n] + oy[m]*Weff1[n] + beff[n]  (f32 VALU, exact fold)
//               + sum_k h[m][k] * W_hh[n][k]                 (v_wmma_f32_16x16x32_bf16)
// h/c live on-chip for all 20 steps; only obs in / final h out touch HBM.
// ---------------------------------------------------------------------------

typedef __attribute__((ext_vector_type(16))) __bf16 v16bf;
typedef __attribute__((ext_vector_type(8)))  float  v8f;

#define TSTEPS 20
#define HIDN   64
#define EMBN   64

__device__ __forceinline__ unsigned short f2bf_bits(float f) {
  union { float f; uint32_t u; } x; x.f = f;
  uint32_t r = x.u + 0x7FFFu + ((x.u >> 16) & 1u);   // round-to-nearest-even
  return (unsigned short)(r >> 16);
}

__device__ __forceinline__ float fast_sigmoid(float x) {
  // 1/(1+2^(-x*log2e)) -> v_exp_f32 + v_rcp_f32
  float e = __builtin_amdgcn_exp2f(-1.4426950408889634f * x);
  return __builtin_amdgcn_rcpf(1.0f + e);
}
__device__ __forceinline__ float fast_tanh(float x) {
  float e = __builtin_amdgcn_exp2f(-2.8853900817779268f * x);
  return 2.0f * __builtin_amdgcn_rcpf(1.0f + e) - 1.0f;
}

// B-fragment (32x16 bf16) load from pre-packed LDS: lane-major 32B rows.
__device__ __forceinline__ v16bf ld_bfrag(const unsigned short* base,
                                          int kt, int nt, int lane) {
  return *(const v16bf*)(base + (((kt * 16 + nt) * 32) + lane) * 16);
}

__global__ __launch_bounds__(256) void lstm_enc_kernel(
    const float* __restrict__ obs,     // [T,B,2]
    const float* __restrict__ W_emb,   // [64,2]
    const float* __restrict__ b_emb,   // [64]
    const float* __restrict__ W_ih,    // [256,64]
    const float* __restrict__ W_hh,    // [256,64]
    const float* __restrict__ b_ih,    // [256]
    const float* __restrict__ b_hh,    // [256]
    float* __restrict__ out,           // [B,64] (final h)
    int B)
{
  // W_hh^T as bf16 B-fragments: [kt(2)][nt(16)][lane(32)][elem(16)]  = 32 KB
  __shared__ __align__(32) unsigned short ldsB[2 * 16 * 32 * 16];
  // per-wave h relayout scratch, A-frag layout: [wave(8)][kt(2)][lane(32)][elem(16)] = 16 KB
  __shared__ __align__(32) unsigned short ldsH[8 * 2 * 32 * 16];
  __shared__ float ldsW0[256], ldsW1[256], ldsBe[256];

  const int tid  = threadIdx.x;
  const int lane = tid & 31;
  const int wave = tid >> 5;
  const int ln   = lane & 15;   // column-within-tile for C/D and B frags
  const int hi   = lane >> 4;   // C/D row-group: rows v (hi=0) or v+8 (hi=1)

  // ---- one-time init: embedding fold + W_hh -> bf16 B-fragment pack ----
  {
    const int n = tid;                          // 256 gate rows, 1 per thread
    float w0 = 0.f, w1 = 0.f, bb = b_ih[n] + b_hh[n];
    #pragma unroll 4
    for (int e = 0; e < EMBN; ++e) {
      float wi = W_ih[n * EMBN + e];
      w0 = fmaf(wi, W_emb[2 * e + 0], w0);
      w1 = fmaf(wi, W_emb[2 * e + 1], w1);
      bb = fmaf(wi, b_emb[e], bb);
    }
    ldsW0[n] = w0; ldsW1[n] = w1; ldsBe[n] = bb;
  }
  for (int idx = tid; idx < 2 * 16 * 32; idx += 256) {
    // B-frag: lanes 0-15 hold K=0..15 of column nt*16+lane; lanes 16-31 hold K=16..31.
    const int kt = idx >> 9;
    const int nt = (idx >> 5) & 15;
    const int l  = idx & 31;
    const int n  = nt * 16 + (l & 15);
    const int kb = kt * 32 + (l >> 4) * 16;
    const float* src = W_hh + n * HIDN + kb;
    unsigned short* dst = ldsB + idx * 16;
    #pragma unroll
    for (int j = 0; j < 16; ++j) dst[j] = f2bf_bits(src[j]);
  }
  __syncthreads();

  // per-lane loop-invariant constants (n = nt*16 + ln for each gate tile)
  float W0c[16], W1c[16], Bec[16];
  #pragma unroll
  for (int nt = 0; nt < 16; ++nt) {
    W0c[nt] = ldsW0[nt * 16 + ln];
    W1c[nt] = ldsW1[nt * 16 + ln];
    Bec[nt] = ldsBe[nt * 16 + ln];
  }

  const int m0 = (blockIdx.x * 8 + wave) * 16;       // this wave's batch rows
  unsigned short* hs = ldsH + wave * (2 * 32 * 16);  // wave-private scratch

  // state: c in f32 C/D layout [cg(4)][v(8)]; h as bf16 A-fragments (K 0..31, 32..63)
  float cst[4][8];
  #pragma unroll
  for (int cg = 0; cg < 4; ++cg)
    #pragma unroll
    for (int v = 0; v < 8; ++v) cst[cg][v] = 0.0f;

  union { unsigned short u; __bf16 b; } bz; bz.u = 0;
  v16bf hfrag0, hfrag1;
  #pragma unroll
  for (int j = 0; j < 16; ++j) { hfrag0[j] = bz.b; hfrag1[j] = bz.b; }

  const float2* obs2 = (const float2*)obs;

  #pragma unroll 1
  for (int t = 0; t < TSTEPS; ++t) {
    // obs for this lane's 8 rows (m = v + 8*hi); uniform per half-wave -> broadcast
    float ox[8], oy[8];
    {
      const float2* op = obs2 + (size_t)t * B + m0 + hi * 8;
      #pragma unroll
      for (int v = 0; v < 8; ++v) { float2 o = op[v]; ox[v] = o.x; oy[v] = o.y; }
    }

    #pragma unroll
    for (int cg = 0; cg < 4; ++cg) {          // h-column group: n_h = cg*16 + ln
      const int ni = cg, nf = cg + 4, ng = cg + 8, no = cg + 12;
      v8f ai, af, ag, ao;
      #pragma unroll
      for (int v = 0; v < 8; ++v) {           // rank-2 x-update + bias, f32-exact
        ai[v] = fmaf(oy[v], W1c[ni], fmaf(ox[v], W0c[ni], Bec[ni]));
        af[v] = fmaf(oy[v], W1c[nf], fmaf(ox[v], W0c[nf], Bec[nf]));
        ag[v] = fmaf(oy[v], W1c[ng], fmaf(ox[v], W0c[ng], Bec[ng]));
        ao[v] = fmaf(oy[v], W1c[no], fmaf(ox[v], W0c[no], Bec[no]));
      }
      // h @ W_hh^T : K=64 -> 2 bf16 WMMAs per gate tile
      ai = __builtin_amdgcn_wmma_f32_16x16x32_bf16(false, hfrag0, false, ld_bfrag(ldsB, 0, ni, lane), (short)0, ai, false, false);
      ai = __builtin_amdgcn_wmma_f32_16x16x32_bf16(false, hfrag1, false, ld_bfrag(ldsB, 1, ni, lane), (short)0, ai, false, false);
      af = __builtin_amdgcn_wmma_f32_16x16x32_bf16(false, hfrag0, false, ld_bfrag(ldsB, 0, nf, lane), (short)0, af, false, false);
      af = __builtin_amdgcn_wmma_f32_16x16x32_bf16(false, hfrag1, false, ld_bfrag(ldsB, 1, nf, lane), (short)0, af, false, false);
      ag = __builtin_amdgcn_wmma_f32_16x16x32_bf16(false, hfrag0, false, ld_bfrag(ldsB, 0, ng, lane), (short)0, ag, false, false);
      ag = __builtin_amdgcn_wmma_f32_16x16x32_bf16(false, hfrag1, false, ld_bfrag(ldsB, 1, ng, lane), (short)0, ag, false, false);
      ao = __builtin_amdgcn_wmma_f32_16x16x32_bf16(false, hfrag0, false, ld_bfrag(ldsB, 0, no, lane), (short)0, ao, false, false);
      ao = __builtin_amdgcn_wmma_f32_16x16x32_bf16(false, hfrag1, false, ld_bfrag(ldsB, 1, no, lane), (short)0, ao, false, false);

      // elementwise LSTM cell + scatter new h into A-frag LDS scratch
      const int ktH  = cg >> 1;                      // k = cg*16+ln -> fragment index
      const int elmH = ln;                           // elem within A-frag
      const int lgrp = 16 * (cg & 1);                // A-frag hi/lo lane group
      #pragma unroll
      for (int v = 0; v < 8; ++v) {
        const int m = v + 8 * hi;                    // row within wave tile
        float iv = fast_sigmoid(ai[v]);
        float fv = fast_sigmoid(af[v]);
        float gv = fast_tanh(ag[v]);
        float ov = fast_sigmoid(ao[v]);
        float cn = fmaf(fv, cst[cg][v], iv * gv);
        cst[cg][v] = cn;
        float hv = ov * fast_tanh(cn);
        hs[(ktH * 32 + (m + lgrp)) * 16 + elmH] = f2bf_bits(hv);
        if (t == TSTEPS - 1)
          out[(size_t)(m0 + m) * HIDN + cg * 16 + ln] = hv;
      }
    }

    // wave-private LDS roundtrip: DS is in-order per wave; make it explicit
    asm volatile("s_wait_dscnt 0" ::: "memory");
    hfrag0 = *(const v16bf*)(hs + (0 * 32 + lane) * 16);
    hfrag1 = *(const v16bf*)(hs + (1 * 32 + lane) * 16);
  }
}

extern "C" void kernel_launch(void* const* d_in, const int* in_sizes, int n_in,
                              void* d_out, int out_size, void* d_ws, size_t ws_size,
                              hipStream_t stream) {
  const float* obs   = (const float*)d_in[0];
  const float* W_emb = (const float*)d_in[1];
  const float* b_emb = (const float*)d_in[2];
  const float* W_ih  = (const float*)d_in[3];
  const float* W_hh  = (const float*)d_in[4];
  const float* b_ih  = (const float*)d_in[5];
  const float* b_hh  = (const float*)d_in[6];
  float* out = (float*)d_out;

  const int B = in_sizes[0] / (2 * TSTEPS);   // 65536
  const int blocks = B / 128;                 // 8 waves * 16 rows per block
  lstm_enc_kernel<<<blocks, 256, 0, stream>>>(obs, W_emb, b_emb, W_ih, W_hh,
                                              b_ih, b_hh, out, B);
}
